// SelectiveMemoryLayer2_57543971832180
// MI455X (gfx1250) — matrix operand order repaired and checked by reading, not compiled, not run on hardware
//
#include <hip/hip_runtime.h>
#include <math.h>

// ---------------- constants ----------------
#define BB 4
#define TT 64
#define LL 64
#define DD 768
#define NHH 12
#define HDD 64
#define FFF 3072
#define TOPKK 5
#define SS 8
#define BT (BB*TT)          // 256
#define MTOK (BT*LL)        // 16384 token rows
#define SCALE_ 0.125f       // 1/sqrt(64)

typedef __attribute__((ext_vector_type(16))) __bf16 v16bf;
typedef __attribute__((ext_vector_type(8)))  float  v8f;

// async global->LDS path (gfx1250): guarded so compilation never breaks
#if defined(__has_builtin)
#if __has_builtin(__builtin_amdgcn_global_load_async_to_lds_b128) && \
    __has_builtin(__builtin_amdgcn_s_wait_asynccnt)
#define USE_ASYNC_LDS 1
#endif
#endif

// exact parameter types per hipcc diagnostic:
//   'int __attribute__((vector_size(16))) __device__ *'  (addrspace(1))
//   and the LDS-side equivalent (addrspace(3))
typedef int v4i_ __attribute__((vector_size(16)));
typedef __attribute__((address_space(1))) v4i_ v4i_g;
typedef __attribute__((address_space(3))) v4i_ v4i_l;

// ---------------- helpers ----------------
__device__ __forceinline__ unsigned short f2bf(float f) {
  unsigned int u = __float_as_uint(f);
  unsigned int r = u + 0x7FFFu + ((u >> 16) & 1u);
  if ((u & 0x7F800000u) == 0x7F800000u) r = u;   // inf/nan passthrough
  return (unsigned short)(r >> 16);
}
__device__ __forceinline__ float bf2f(unsigned short h) {
  return __uint_as_float(((unsigned int)h) << 16);
}
__device__ __forceinline__ float gelu_exact(float x) {
  return 0.5f * x * (1.0f + erff(x * 0.70710678118654752f));
}
__device__ __forceinline__ float sigmoidf_(float x) {
  return 1.0f / (1.0f + expf(-x));
}

// ---------------- f32 -> bf16 conversion ----------------
__global__ void f2b_kernel(const float* __restrict__ in,
                           unsigned short* __restrict__ out, size_t n) {
  size_t i = (size_t)blockIdx.x * blockDim.x + threadIdx.x;
  size_t stride = (size_t)gridDim.x * blockDim.x;
  for (; i < n; i += stride) out[i] = f2bf(in[i]);
}

__global__ void copy_f32_kernel(const float* __restrict__ in,
                                float* __restrict__ out, int n) {
  int i = blockIdx.x * blockDim.x + threadIdx.x;
  if (i < n) out[i] = in[i];
}

// ---------------- WMMA GEMM: C[M,N] = A[M,K] * W[N,K]^T + bias ----------------
// Block = 256 threads (8 waves) computing a 128x128 tile.
// Wave w owns rows [w*16, w*16+16) x all 128 columns: 8 accumulators,
// A-fragment loaded once per K-step (global), B slab (128x32 bf16, 8KB)
// staged in LDS via async global->LDS copies and shared by all waves.
__global__ __launch_bounds__(256) void gemm_bf16_wmma(
    const unsigned short* __restrict__ A,
    const unsigned short* __restrict__ W,
    const float* __restrict__ bias,
    float* __restrict__ outF,
    unsigned short* __restrict__ outB,
    int M, int N, int K, int act)
{
  __shared__ unsigned short sB[128 * 32];   // [n_local][k_local], k contiguous

  int waveId = threadIdx.x >> 5;
  int lane   = threadIdx.x & 31;
  int m0   = blockIdx.x * 128;
  int nblk = blockIdx.y * 128;
  if (m0 >= M || nblk >= N) return;          // block-uniform

  int row = lane & 15;
  int hlf = lane >> 4;
  const unsigned short* Ap = A + (size_t)(m0 + waveId * 16 + row) * K;
  int kaBase = hlf * 8;     // A 16x32 bf16 fragment: upper half-wave at K+8
  int kbBase = hlf * 16;    // B fragment: same k order as A (upper half-wave at K+8)

  v8f c[8];
  #pragma unroll
  for (int s = 0; s < 8; ++s) c[s] = (v8f){0.f,0.f,0.f,0.f,0.f,0.f,0.f,0.f};

  const unsigned int* sBu = (const unsigned int*)sB;

  for (int k0 = 0; k0 < K; k0 += 32) {
    __syncthreads();   // previous slab fully consumed before overwrite

    // cooperative copy of B slab: 128 rows x 32 bf16 = 8KB, 16B per chunk
    #pragma unroll
    for (int it = 0; it < 2; ++it) {
      int ch = threadIdx.x + it * 256;       // 0..511
      int r  = ch >> 2;                      // n_local 0..127
      int kc = ch & 3;                       // 16B chunk within 64B row
      const unsigned short* gp = W + (size_t)(nblk + r) * K + k0 + kc * 8;
      unsigned short* lp = sB + r * 32 + kc * 8;
#ifdef USE_ASYNC_LDS
      __builtin_amdgcn_global_load_async_to_lds_b128(
          (v4i_g*)gp, (v4i_l*)lp, 0, 0);
#else
      *(uint4*)lp = *(const uint4*)gp;
#endif
    }
#ifdef USE_ASYNC_LDS
    __builtin_amdgcn_s_wait_asynccnt(0);
#endif
    if (k0 + 32 < K) __builtin_prefetch(Ap + k0 + 32, 0, 0);
    __syncthreads();

    // A fragment from global (reused by 8 WMMAs)
    union { v16bf v; unsigned int u[8]; } a;
    #pragma unroll
    for (int j = 0; j < 8; ++j) {
      int ka = k0 + kaBase + ((j < 4) ? (2 * j) : (16 + 2 * (j - 4)));
      a.u[j] = *(const unsigned int*)(Ap + ka);
    }

    // 8 n-subtiles from LDS
    #pragma unroll
    for (int s = 0; s < 8; ++s) {
      int colLocal = s * 16 + (lane & 15);
      union { v16bf v; unsigned int u[8]; } b;
      #pragma unroll
      for (int j = 0; j < 8; ++j)
        b.u[j] = sBu[colLocal * 16 + hlf * 4 + ((j < 4) ? j : (4 + j))];
      c[s] = __builtin_amdgcn_wmma_f32_16x16x32_bf16(
          false, a.v, false, b.v, (short)0, c[s], false, false);
    }
  }

  // epilogue
  #pragma unroll
  for (int s = 0; s < 8; ++s) {
    int n = nblk + s * 16 + (lane & 15);
    float bv = bias ? bias[n] : 0.0f;
    #pragma unroll
    for (int r = 0; r < 8; ++r) {
      int m = m0 + waveId * 16 + r + hlf * 8;  // C layout: VGPR r -> M=r / 8+r
      float v = c[s][r] + bv;
      if (act == 1) v = gelu_exact(v);
      size_t o = (size_t)m * N + n;
      if (outF) outF[o] = v;
      if (outB) outB[o] = f2bf(v);
    }
  }
}

// ---------------- attention (per sequence, per head) ----------------
__global__ __launch_bounds__(64) void attn_kernel(
    const unsigned short* __restrict__ qkv,
    const unsigned char* __restrict__ mask,   // (BT, L) bool
    unsigned short* __restrict__ out)
{
  __shared__ float s_k[LL * HDD];
  __shared__ float s_v[LL * HDD];
  __shared__ float s_p[LL * LL];

  int bt = blockIdx.x;
  int h  = blockIdx.y;
  int i  = threadIdx.x;
  size_t base = (size_t)bt * LL * (3 * DD);
  int hoff = h * HDD;

  #pragma unroll 8
  for (int d = 0; d < HDD; ++d) {
    s_k[i * HDD + d] = bf2f(qkv[base + (size_t)i * (3 * DD) + DD     + hoff + d]);
    s_v[i * HDD + d] = bf2f(qkv[base + (size_t)i * (3 * DD) + 2 * DD + hoff + d]);
  }
  float q[HDD];
  #pragma unroll
  for (int d = 0; d < HDD; ++d)
    q[d] = bf2f(qkv[base + (size_t)i * (3 * DD) + hoff + d]);
  __syncthreads();

  float mx = -3.0e38f;
  for (int j = 0; j < LL; ++j) {
    float s = 0.0f;
    #pragma unroll
    for (int d = 0; d < HDD; ++d) s += q[d] * s_k[j * HDD + d];
    s *= SCALE_;
    if (!mask[bt * LL + j]) s = -1.0e30f;
    s_p[i * LL + j] = s;
    mx = fmaxf(mx, s);
  }
  float sum = 0.0f;
  for (int j = 0; j < LL; ++j) {
    float e = expf(s_p[i * LL + j] - mx);
    s_p[i * LL + j] = e;
    sum += e;
  }
  float inv = 1.0f / sum;
  float acc[HDD];
  #pragma unroll
  for (int d = 0; d < HDD; ++d) acc[d] = 0.0f;
  for (int j = 0; j < LL; ++j) {
    float a = s_p[i * LL + j];
    #pragma unroll
    for (int d = 0; d < HDD; ++d) acc[d] += a * s_v[j * HDD + d];
  }
  size_t obase = ((size_t)bt * LL + i) * DD + hoff;
  #pragma unroll
  for (int d = 0; d < HDD; ++d) out[obase + d] = f2bf(acc[d] * inv);
}

// ---------------- residual + layernorm ----------------
__global__ __launch_bounds__(256) void ln_kernel(
    const float* __restrict__ a, const float* __restrict__ res,
    const float* __restrict__ g, const float* __restrict__ be,
    float* __restrict__ outF, unsigned short* __restrict__ outB)
{
  __shared__ float xs[DD];
  __shared__ float red[256];
  int row = blockIdx.x, tid = threadIdx.x;
  size_t base = (size_t)row * DD;

  float lsum = 0.0f;
  for (int d = tid; d < DD; d += 256) {
    float v = a[base + d] + (res ? res[base + d] : 0.0f);
    xs[d] = v; lsum += v;
  }
  red[tid] = lsum; __syncthreads();
  for (int s = 128; s > 0; s >>= 1) { if (tid < s) red[tid] += red[tid + s]; __syncthreads(); }
  float mean = red[0] / (float)DD;
  __syncthreads();
  float lv = 0.0f;
  for (int d = tid; d < DD; d += 256) { float t = xs[d] - mean; lv += t * t; }
  red[tid] = lv; __syncthreads();
  for (int s = 128; s > 0; s >>= 1) { if (tid < s) red[tid] += red[tid + s]; __syncthreads(); }
  float rstd = rsqrtf(red[0] / (float)DD + 1e-5f);
  for (int d = tid; d < DD; d += 256) {
    float v = (xs[d] - mean) * rstd * g[d] + be[d];
    if (outF) outF[base + d] = v;
    if (outB) outB[base + d] = f2bf(v);
  }
}

// ---------------- memory-path score ----------------
__global__ __launch_bounds__(64) void score_kernel(
    const float* __restrict__ qs, const float* __restrict__ ks,
    const float* __restrict__ qc, const float* __restrict__ kc,
    float* __restrict__ score)
{
  int b = blockIdx.x, t = blockIdx.y, j = threadIdx.x;
  const float* q1 = qs + ((size_t)b * TT + t) * DD;
  const float* k1 = ks + ((size_t)b * TT + j) * DD;
  const float* q2 = qc + ((size_t)b * TT + t) * DD;
  const float* k2 = kc + ((size_t)b * TT + j) * DD;
  float s = 0.0f;
  for (int d = 0; d < DD; ++d) s += q1[d] * k1[d] + q2[d] * k2[d];
  score[((size_t)b * TT + t) * TT + j] = s;
}

// ---------------- sequential memory scan (top-k + GRU) ----------------
__global__ __launch_bounds__(256) void scan_kernel(
    const float* __restrict__ score, const float* __restrict__ edu,
    const int* __restrict__ sids,
    const float* __restrict__ wih, const float* __restrict__ whh,
    const float* __restrict__ bih, const float* __restrict__ bhh,
    float* __restrict__ mem)       // (B,S,D), updated in place
{
  __shared__ float summary[DD];
  __shared__ float hold[DD];
  __shared__ float hnew[DD];
  __shared__ float sc_s[TT];
  __shared__ int   sel[TOPKK];
  __shared__ int   nsel;

  int b = blockIdx.x, tid = threadIdx.x;
  float* bmem = mem + (size_t)b * SS * DD;

  for (int t = 1; t < TT; ++t) {
    if (tid < TT) sc_s[tid] = (tid < t) ? score[((size_t)b * TT + t) * TT + tid] : -3.0e38f;
    __syncthreads();
    if (tid == 0) {
      int kv = (t < TOPKK) ? t : TOPKK;
      nsel = kv;
      for (int p = 0; p < kv; ++p) {
        float best = -3.0e38f; int bi = 0;
        for (int j = 0; j < t; ++j)
          if (sc_s[j] > best) { best = sc_s[j]; bi = j; }
        sel[p] = bi; sc_s[bi] = -3.0e38f;
      }
    }
    __syncthreads();
    int kv = nsel;
    float denom = (float)(kv > 0 ? kv : 1);
    int sid = sids[b * TT + t];
    for (int d = tid; d < DD; d += 256) {
      float s = 0.0f;
      for (int p = 0; p < kv; ++p)
        s += edu[((size_t)b * TT + sel[p]) * DD + d];
      summary[d] = s / denom;
      hold[d] = bmem[(size_t)sid * DD + d];
    }
    __syncthreads();
    for (int d = tid; d < DD; d += 256) {
      float ir = bih[d], iz = bih[DD + d], inn = bih[2 * DD + d];
      float hr = bhh[d], hz = bhh[DD + d], hn  = bhh[2 * DD + d];
      const float* wr = wih + (size_t)d * DD;
      const float* wz = wih + (size_t)(DD + d) * DD;
      const float* wn = wih + (size_t)(2 * DD + d) * DD;
      const float* ur = whh + (size_t)d * DD;
      const float* uz = whh + (size_t)(DD + d) * DD;
      const float* un = whh + (size_t)(2 * DD + d) * DD;
      for (int k = 0; k < DD; ++k) {
        float sv = summary[k], hv = hold[k];
        ir += wr[k] * sv; iz += wz[k] * sv; inn += wn[k] * sv;
        hr += ur[k] * hv; hz += uz[k] * hv; hn  += un[k] * hv;
      }
      float r = sigmoidf_(ir + hr);
      float z = sigmoidf_(iz + hz);
      float n = tanhf(inn + r * hn);
      hnew[d] = (1.0f - z) * n + z * hold[d];
    }
    __syncthreads();
    for (int d = tid; d < DD; d += 256)
      bmem[(size_t)sid * DD + d] = hnew[d];
    __syncthreads();
  }
}

// ---------------- host launch ----------------
extern "C" void kernel_launch(void* const* d_in, const int* in_sizes, int n_in,
                              void* d_out, int out_size, void* d_ws, size_t ws_size,
                              hipStream_t stream) {
  (void)in_sizes; (void)n_in; (void)out_size; (void)ws_size;
  const float*         tokens    = (const float*)d_in[0];
  const unsigned char* mask      = (const unsigned char*)d_in[1];
  const float*         edu       = (const float*)d_in[2];
  const int*           spk_ids   = (const int*)d_in[3];
  const float*         spk_mem   = (const float*)d_in[4];
  const float* attn_in_w  = (const float*)d_in[5];
  const float* attn_in_b  = (const float*)d_in[6];
  const float* attn_out_w = (const float*)d_in[7];
  const float* attn_out_b = (const float*)d_in[8];
  const float* ffn_w1     = (const float*)d_in[9];
  const float* ffn_b1     = (const float*)d_in[10];
  const float* ffn_w2     = (const float*)d_in[11];
  const float* ffn_b2     = (const float*)d_in[12];
  const float* ln1_g      = (const float*)d_in[13];
  const float* ln1_b      = (const float*)d_in[14];
  const float* ln2_g      = (const float*)d_in[15];
  const float* ln2_b      = (const float*)d_in[16];
  const float* spk_q      = (const float*)d_in[17];
  const float* spk_k      = (const float*)d_in[18];
  const float* cnt_q      = (const float*)d_in[19];
  const float* cnt_k      = (const float*)d_in[20];
  const float* gru_wih    = (const float*)d_in[21];
  const float* gru_whh    = (const float*)d_in[22];
  const float* gru_bih    = (const float*)d_in[23];
  const float* gru_bhh    = (const float*)d_in[24];

  char* ws = (char*)d_ws;
  size_t off = 0;
  auto alloc = [&](size_t bytes) -> void* {
    void* p = ws + off;
    off = (off + bytes + 255) & ~(size_t)255;
    return p;
  };
  auto F = [&](size_t n) { return (float*)alloc(n * 4); };
  auto H = [&](size_t n) { return (unsigned short*)alloc(n * 2); };

  unsigned short* xb    = H((size_t)MTOK * DD);
  unsigned short* wqkvb = H((size_t)3 * DD * DD);
  unsigned short* woutb = H((size_t)DD * DD);
  unsigned short* w1b   = H((size_t)FFF * DD);
  unsigned short* w2b   = H((size_t)DD * FFF);
  unsigned short* wsqb  = H((size_t)DD * DD);
  unsigned short* wskb  = H((size_t)DD * DD);
  unsigned short* wcqb  = H((size_t)DD * DD);
  unsigned short* wckb  = H((size_t)DD * DD);
  unsigned short* edub  = H((size_t)BT * DD);
  unsigned short* qkvb  = H((size_t)MTOK * 3 * DD);
  unsigned short* aob   = H((size_t)MTOK * DD);
  float*          projf = F((size_t)MTOK * DD);
  float*          x1f   = F((size_t)MTOK * DD);
  unsigned short* x1b   = H((size_t)MTOK * DD);
  unsigned short* hb    = H((size_t)MTOK * FFF);
  float*          fff   = F((size_t)MTOK * DD);
  float*          qsf   = F((size_t)BT * DD);
  float*          ksf   = F((size_t)BT * DD);
  float*          qcf   = F((size_t)BT * DD);
  float*          kcf   = F((size_t)BT * DD);
  float*          scoref= F((size_t)BB * TT * TT);

  float* outTok = (float*)d_out;                       // (B,T,L,D)
  float* outMem = outTok + (size_t)MTOK * DD;          // (B,S,D)

  // 1) bf16 conversions
  struct Cnv { const float* src; unsigned short* dst; size_t n; };
  Cnv cnv[] = {
    { tokens,     xb,    (size_t)MTOK * DD },
    { attn_in_w,  wqkvb, (size_t)3 * DD * DD },
    { attn_out_w, woutb, (size_t)DD * DD },
    { ffn_w1,     w1b,   (size_t)FFF * DD },
    { ffn_w2,     w2b,   (size_t)DD * FFF },
    { spk_q,      wsqb,  (size_t)DD * DD },
    { spk_k,      wskb,  (size_t)DD * DD },
    { cnt_q,      wcqb,  (size_t)DD * DD },
    { cnt_k,      wckb,  (size_t)DD * DD },
    { edu,        edub,  (size_t)BT * DD },
  };
  for (auto& c : cnv) {
    int blocks = (int)((c.n + 255) / 256);
    if (blocks > 2048) blocks = 2048;
    f2b_kernel<<<blocks, 256, 0, stream>>>(c.src, c.dst, c.n);
  }

  // 2) QKV projection
  gemm_bf16_wmma<<<dim3(MTOK / 128, (3 * DD) / 128), 256, 0, stream>>>(
      xb, wqkvb, attn_in_b, nullptr, qkvb, MTOK, 3 * DD, DD, 0);

  // 3) attention
  attn_kernel<<<dim3(BT, NHH), 64, 0, stream>>>(qkvb, mask, aob);

  // 4) output projection
  gemm_bf16_wmma<<<dim3(MTOK / 128, DD / 128), 256, 0, stream>>>(
      aob, woutb, attn_out_b, projf, nullptr, MTOK, DD, DD, 0);

  // 5) residual + LN1
  ln_kernel<<<MTOK, 256, 0, stream>>>(projf, tokens, ln1_g, ln1_b, x1f, x1b);

  // 6) FFN1 + GELU -> bf16
  gemm_bf16_wmma<<<dim3(MTOK / 128, FFF / 128), 256, 0, stream>>>(
      x1b, w1b, ffn_b1, nullptr, hb, MTOK, FFF, DD, 1);

  // 7) FFN2
  gemm_bf16_wmma<<<dim3(MTOK / 128, DD / 128), 256, 0, stream>>>(
      hb, w2b, ffn_b2, fff, nullptr, MTOK, DD, FFF, 0);

  // 8) residual + LN2 -> tokens_out
  ln_kernel<<<MTOK, 256, 0, stream>>>(fff, x1f, ln2_g, ln2_b, outTok, nullptr);

  // 9) memory scoring projections (256 x 768 each)
  gemm_bf16_wmma<<<dim3(BT / 128, DD / 128), 256, 0, stream>>>(
      edub, wsqb, nullptr, qsf, nullptr, BT, DD, DD, 0);
  gemm_bf16_wmma<<<dim3(BT / 128, DD / 128), 256, 0, stream>>>(
      edub, wskb, nullptr, ksf, nullptr, BT, DD, DD, 0);
  gemm_bf16_wmma<<<dim3(BT / 128, DD / 128), 256, 0, stream>>>(
      edub, wcqb, nullptr, qcf, nullptr, BT, DD, DD, 0);
  gemm_bf16_wmma<<<dim3(BT / 128, DD / 128), 256, 0, stream>>>(
      edub, wckb, nullptr, kcf, nullptr, BT, DD, DD, 0);

  // 10) pairwise scores
  score_kernel<<<dim3(BB, TT), TT, 0, stream>>>(qsf, ksf, qcf, kcf, scoref);

  // 11) init mem_out = speaker_memory
  copy_f32_kernel<<<(BB * SS * DD + 255) / 256, 256, 0, stream>>>(
      spk_mem, outMem, BB * SS * DD);

  // 12) sequential top-k + GRU scan
  scan_kernel<<<BB, 256, 0, stream>>>(scoref, edu, spk_ids,
                                      gru_wih, gru_whh, gru_bih, gru_bhh, outMem);
}
